// SelfAttention_46626164965493
// MI455X (gfx1250) — matrix operand order repaired and checked by reading, not compiled, not run on hardware
//
#include <hip/hip_runtime.h>

// Problem constants (match reference)
#define BB 16
#define CC 128
#define C8 16
#define NN 4096   // H*W
#define NT 64     // tile of pixels / keys per block

typedef __attribute__((ext_vector_type(16))) _Float16 v16h;
typedef __attribute__((ext_vector_type(8)))  _Float16 v8h;
typedef __attribute__((ext_vector_type(8)))  float    v8f;

union V16U { v16h v; v8h h[2]; };

// ---------------------------------------------------------------------------
// Kernel 0: zero the per-batch max||k||^2 slots
// ---------------------------------------------------------------------------
__global__ void init_maxk_kernel(unsigned* __restrict__ maxkn2) {
  if (threadIdx.x < BB) maxkn2[threadIdx.x] = 0u;
}

// ---------------------------------------------------------------------------
// Kernel 1: Q/K projection (1x1 conv) -> f16, rows padded from K=16 to K=32.
// Also reduces max_j ||k_j||^2 per batch (integer atomicMax: bit order ==
// float order for non-negative floats).
// ---------------------------------------------------------------------------
__global__ __launch_bounds__(64) void proj_qk_kernel(
    const float* __restrict__ x,  const float* __restrict__ wq,
    const float* __restrict__ bq, const float* __restrict__ wk,
    const float* __restrict__ bk,
    _Float16* __restrict__ Qh, _Float16* __restrict__ Kh,
    unsigned* __restrict__ maxkn2)
{
  __shared__ float s_wq[C8 * CC];
  __shared__ float s_wk[C8 * CC];
  const int tid = threadIdx.x;
  for (int i = tid; i < C8 * CC; i += 64) { s_wq[i] = wq[i]; s_wk[i] = wk[i]; }
  __syncthreads();

  const int b = blockIdx.y;
  const int n = blockIdx.x * NT + tid;

  float qa[C8], ka[C8];
  #pragma unroll
  for (int o = 0; o < C8; ++o) { qa[o] = bq[o]; ka[o] = bk[o]; }

  const float* xp = x + (size_t)b * CC * NN + n;
  for (int c = 0; c < CC; ++c) {
    float xv = xp[(size_t)c * NN];                 // coalesced in n
    #pragma unroll
    for (int o = 0; o < C8; ++o) {
      qa[o] = fmaf(xv, s_wq[o * CC + c], qa[o]);   // uniform addr -> broadcast
      ka[o] = fmaf(xv, s_wk[o * CC + c], ka[o]);
    }
  }
  _Float16* qrow = Qh + ((size_t)b * NN + n) * 32;
  _Float16* krow = Kh + ((size_t)b * NN + n) * 32;
  float kn2 = 0.f;
  #pragma unroll
  for (int o = 0; o < C8; ++o) {
    qrow[o] = (_Float16)qa[o]; qrow[o + 16] = (_Float16)0.f;
    krow[o] = (_Float16)ka[o]; krow[o + 16] = (_Float16)0.f;
    kn2 = fmaf(ka[o], ka[o], kn2);
  }
  // wave-level pre-reduction, then one atomic per wave
  #pragma unroll
  for (int d = 1; d < 32; d <<= 1) kn2 = fmaxf(kn2, __shfl_xor(kn2, d, 32));
  if ((tid & 31) == 0) atomicMax(&maxkn2[b], __float_as_uint(kn2));
}

// ---------------------------------------------------------------------------
// Kernel 2: V projection, written directly in WMMA B-operand tile order:
// VtB[b][chunk(64 keys)][ (ks*2+hi)*8+oc ][ j(16) ][ t(16) ]  (f16)
// ---------------------------------------------------------------------------
__global__ __launch_bounds__(256) void proj_v_kernel(
    const float* __restrict__ x, const float* __restrict__ wv,
    const float* __restrict__ bv, _Float16* __restrict__ VtB)
{
  __shared__ float s_x[CC * NT];   // [c][n], 32 KB
  const int tid = threadIdx.x;
  const int b  = blockIdx.y;
  const int n0 = blockIdx.x * NT;

  for (int i = tid; i < CC * NT; i += 256) {
    int c = i / NT, nn = i % NT;
    s_x[i] = x[((size_t)b * CC + c) * NN + n0 + nn];   // coalesced in nn
  }
  __syncthreads();

  const int nn  = tid & 63;     // key within chunk
  const int grp = tid >> 6;     // wave-uniform (0..3)
  const int cb  = grp * 32;

  float acc[32];
  #pragma unroll
  for (int j = 0; j < 32; ++j) acc[j] = bv[cb + j];

  for (int c = 0; c < CC; ++c) {
    float xv = s_x[c * NT + nn];
    #pragma unroll
    for (int j = 0; j < 32; ++j)
      acc[j] = fmaf(xv, wv[(cb + j) * CC + c], acc[j]);  // wave-uniform -> s_load
  }

  const int ks = nn >> 5, h2 = (nn >> 4) & 1, t = nn & 15;
  _Float16* blob = VtB + ((size_t)(b * (NN / NT) + blockIdx.x)) * (NT * CC)
                 + (ks * 2 + h2) * 2048 + t;
  #pragma unroll
  for (int j = 0; j < 32; ++j) {
    int ch = cb + j;
    blob[((ch >> 4) * 16 + (ch & 15)) * 16] = (_Float16)acc[j];
  }
}

// ---------------------------------------------------------------------------
// Kernel 3: attention with fixed softmax shift M_i = ||q_i||*max||k|| and
// row sums accumulated by a ones-column WMMA. Inner loop: no cross-lane ops,
// no barriers, no accumulator rescale. 22 wmma per 64-key chunk per wave.
// Softmax pre-op: one dual-FMA + raw v_exp_f32 per element.
// ---------------------------------------------------------------------------
__global__ __launch_bounds__(128) void attn_kernel(
    const _Float16* __restrict__ Qh, const _Float16* __restrict__ Kh,
    const _Float16* __restrict__ VtB, const float* __restrict__ x,
    const float* __restrict__ gamma, float* __restrict__ out,
    const unsigned* __restrict__ maxkn2)
{
  // loop phase: s_p (4 waves x 16 x 72 halves, wave-private, padded)
  // final phase: s_o (64 x 132 f32, padded) -- overlaps s_p
  __shared__ __align__(16) unsigned char smem[64 * 132 * 4];   // 33792 B
  _Float16* s_p = (_Float16*)smem;
  float*    s_o = (float*)smem;

  const int tid  = threadIdx.x;
  const int lane = tid & 31;
  const int w    = tid >> 5;       // wave id 0..3
  const int b    = blockIdx.y;
  const int i0   = blockIdx.x * NT;
  const int m    = lane & 15;
  const int hi   = lane >> 4;

  // A-operand for Q strip (rows i0+16w+m), contiguous k = 16*hi + s layout
  V16U a_q;
  {
    const _Float16* qrow = Qh + ((size_t)b * NN + (i0 + 16 * w + m)) * 32;
    a_q.h[0] = *(const v8h*)(qrow + hi * 16);
    a_q.h[1] = *(const v8h*)(qrow + hi * 16 + 8);
  }

  // Per-row fused softmax shift (C-domain rows r+8*hi), computed once:
  // M2[r] = -||q_row||*max||k||*log2(e)  (upper bound via Cauchy-Schwarz,
  // so the exp2 argument is always <= 0; raw v_exp_f32 is safe).
  const float LOG2E = 1.44269504088896f;
  float M2[8];
  {
    float qn2 = 0.f;
    #pragma unroll
    for (int t = 0; t < 16; ++t) {
      float qv = (float)a_q.v[t];
      qn2 = fmaf(qv, qv, qn2);
    }
    qn2 += __shfl_xor(qn2, 16, 32);
    const float maxk = sqrtf(__uint_as_float(maxkn2[b]));
    #pragma unroll
    for (int r = 0; r < 8; ++r)
      M2[r] = -sqrtf(__shfl(qn2, 8 * hi + r, 32)) * maxk * LOG2E;
  }

  // constant all-ones B operand for row-sum wmma
  v16h vone;
  #pragma unroll
  for (int t = 0; t < 16; ++t) vone[t] = (_Float16)1.0f;

  v8f acc[8];                  // O strip 16x128 f32 (8 col tiles)
  v8f lacc;                    // row sums of P (every column identical)
  #pragma unroll
  for (int oc = 0; oc < 8; ++oc)
    #pragma unroll
    for (int r = 0; r < 8; ++r) acc[oc][r] = 0.f;
  #pragma unroll
  for (int r = 0; r < 8; ++r) lacc[r] = 0.f;

  const size_t kbase = (size_t)b * NN * 32;             // halves
  const size_t vbase = (size_t)b * (NN / NT) * (NT * CC);

  for (int j0 = 0; j0 < NN; j0 += NT) {
    const _Float16* vblob = VtB + vbase + (size_t)(j0 >> 6) * (NT * CC);

    if (j0 + NT < NN) {          // pull next chunk toward L2/L0
      __builtin_prefetch(Kh + kbase + (size_t)(j0 + NT) * 32 + lane * 64, 0, 1);
      __builtin_prefetch(vblob + (NT * CC) + lane * 256, 0, 1);
    }

    // ---- S = Q K^T over 64 keys: 4 wmma tiles (K=32, zero-padded) ----
    v8f s[4];
    #pragma unroll
    for (int jt = 0; jt < 4; ++jt) {
      const _Float16* krow = Kh + kbase + (size_t)(j0 + jt * 16 + m) * 32;
      v16h bk = *(const v16h*)(krow + hi * 16);
      v8f  z  = {};
      s[jt] = __builtin_amdgcn_wmma_f32_16x16x32_f16(
          false, a_q.v, false, bk, (short)0, z, false, false);
    }

    // ---- P = exp2(S*log2e + M2) -> wave-private LDS (rows of 72 halves) ----
    // raw hardware exp2: one dual-FMA + one v_exp_f32 per element
    #pragma unroll
    for (int r = 0; r < 8; ++r) {
      const int row = (w * 16 + r + 8 * hi) * 72;
      #pragma unroll
      for (int jt = 0; jt < 4; ++jt)
        s_p[row + jt * 16 + m] =
            (_Float16)__builtin_amdgcn_exp2f(fmaf(s[jt][r], LOG2E, M2[r]));
    }

    // ---- O += P * V  and  l += P * 1 : 2 k-steps ----
    #pragma unroll
    for (int ks = 0; ks < 2; ++ks) {
      V16U a_p;
      const _Float16* prow = s_p + (w * 16 + m) * 72 + ks * 32;
      a_p.h[0] = *(const v8h*)(prow + hi * 16);
      a_p.h[1] = *(const v8h*)(prow + hi * 16 + 8);
      lacc = __builtin_amdgcn_wmma_f32_16x16x32_f16(
          false, a_p.v, false, vone, (short)0, lacc, false, false);
      #pragma unroll
      for (int oc = 0; oc < 8; ++oc) {
        v16h bvv = *(const v16h*)(vblob + ((ks * 2 + hi) * 8 + oc) * 256 + m * 16);
        acc[oc] = __builtin_amdgcn_wmma_f32_16x16x32_f16(
            false, a_p.v, false, bvv, (short)0, acc[oc], false, false);
      }
    }
  }

  __syncthreads();   // s_o overlaps other waves' s_p regions

  // normalize rows and stage for coalesced output: s_o[row_in_tile][ch]
  #pragma unroll
  for (int r = 0; r < 8; ++r) {
    float inv = 1.0f / lacc[r];
    int row = w * 16 + r + 8 * hi;
    #pragma unroll
    for (int oc = 0; oc < 8; ++oc)
      s_o[row * 132 + oc * 16 + m] = acc[oc][r] * inv;
  }
  __syncthreads();

  const float g = gamma[0];
  for (int idx = tid; idx < CC * NT; idx += 128) {
    int c = idx >> 6;
    int t = idx & 63;
    size_t gi = ((size_t)b * CC + c) * NN + i0 + t;   // coalesced in t
    out[gi] = g * s_o[t * 132 + c] + x[gi];
  }
}

// ---------------------------------------------------------------------------
extern "C" void kernel_launch(void* const* d_in, const int* in_sizes, int n_in,
                              void* d_out, int out_size, void* d_ws, size_t ws_size,
                              hipStream_t stream) {
  const float* x  = (const float*)d_in[0];
  const float* wq = (const float*)d_in[1];
  const float* bq = (const float*)d_in[2];
  const float* wk = (const float*)d_in[3];
  const float* bk = (const float*)d_in[4];
  const float* wv = (const float*)d_in[5];
  const float* bv = (const float*)d_in[6];
  const float* gm = (const float*)d_in[7];
  float* out = (float*)d_out;

  // workspace: Qh 4MB | Kh 4MB | VtB 16MB | maxkn2 (16 u32)
  _Float16* Qh  = (_Float16*)d_ws;
  _Float16* Kh  = Qh + (size_t)BB * NN * 32;
  _Float16* VtB = Kh + (size_t)BB * NN * 32;
  unsigned* mk  = (unsigned*)(VtB + (size_t)BB * NN * CC);

  dim3 grid(NN / NT, BB);
  init_maxk_kernel<<<1, 32, 0, stream>>>(mk);
  proj_qk_kernel<<<grid,  64, 0, stream>>>(x, wq, bq, wk, bk, Qh, Kh, mk);
  proj_v_kernel <<<grid, 256, 0, stream>>>(x, wv, bv, VtB);
  attn_kernel   <<<grid, 128, 0, stream>>>(Qh, Kh, VtB, x, gm, out, mk);
}
